// SelfAttention_61607010894413
// MI455X (gfx1250) — matrix operand order repaired and checked by reading, not compiled, not run on hardware
//
#include <hip/hip_runtime.h>
#include <hip/hip_bf16.h>

// ---------------------------------------------------------------------------
// Self-attention block for MI455X (gfx1250).
// All matmuls via v_wmma_f32_16x16x32_bf16 (bf16 operands, fp32 accumulate).
// GEMM: double-buffered LDS pipeline using ASYNCcnt-tracked
// global_load_async_to_lds_b128 with a partial s_wait_asynccnt so tile t+1
// copies overlap tile t WMMAs. Wave tile 32x64 (8 WMMA per k-step) to raise
// the WMMA:ds ratio. All fp32->bf16 conversion hoisted to a pre-pass.
//
// Shapes: B=4, T=1024, C=2048, H=32, KV=8, D=64.
// ---------------------------------------------------------------------------

#define B_   4
#define T_   1024
#define C_   2048
#define H_   32
#define KV_  8
#define D_   64
#define KVD_ 512
#define MTOT (B_ * T_)   // 4096

#define USE_ASYNC_LDS 1

typedef __attribute__((ext_vector_type(16))) __bf16 v16bf;
typedef __attribute__((ext_vector_type(8)))  float  v8f;

__device__ inline v8f wmma_bf16(v16bf a, v16bf b, v8f c) {
  return __builtin_amdgcn_wmma_f32_16x16x32_bf16(false, a, false, b, (short)0, c,
                                                 false, false);
}

// 16-byte global -> LDS copy, ASYNCcnt-tracked (CDNA5 ISA).
// VDST = LDS byte offset (low 32 bits of a generic pointer to __shared__).
__device__ inline void cp_async16(void* ldsp, const void* gp) {
#if USE_ASYNC_LDS
  unsigned lds_off = (unsigned)(size_t)ldsp;
  asm volatile("global_load_async_to_lds_b128 %0, %1, off"
               :: "v"(lds_off), "v"(gp) : "memory");
#else
  *(float4*)ldsp = *(const float4*)gp;
#endif
}

// Wait until ALL async copies landed.
__device__ inline void cp_async_wait_all() {
#if USE_ASYNC_LDS
  asm volatile("s_wait_asynccnt 0x0" ::: "memory");
#endif
}

// Allow the 8 most recent async copies (next tile, 8 per thread) to remain
// in flight; ASYNCcnt loads complete in order, so this means the previous
// tile's copies have landed.
__device__ inline void cp_async_wait_keep8() {
#if USE_ASYNC_LDS
  asm volatile("s_wait_asynccnt 0x8" ::: "memory");
#endif
}

// ---------------------------------------------------------------------------
// Fragment layouts used here (contiguous k, not the ISA table order; A and B match):
// A (16x32): lane L -> row M = L&15; elem e -> K = e + 16*(L>=16)
// B (32x16): lane L -> col N = L&15; elem e -> K = e + 16*(L>=16)
// ---------------------------------------------------------------------------
__device__ inline v16bf load_a_frag(const __bf16* __restrict__ base, int ld,
                                    int row0, int k0, int lane) {
  int m    = lane & 15;
  int koff = (lane & 16) ? 16 : 0;
  const __bf16* p = base + (size_t)(row0 + m) * ld + (k0 + koff);
  v16bf a;
#pragma unroll
  for (int e = 0; e < 8; ++e) { a[e] = p[e]; a[e + 8] = p[e + 8]; }
  return a;
}

// B fragment where the source is "K-fast": row n holds contiguous K elements.
__device__ inline v16bf load_b_frag_kfast(const __bf16* __restrict__ base,
                                          int ld, int n0, int k0, int lane) {
  int n = n0 + (lane & 15);
  const __bf16* p = base + (size_t)n * ld + k0 + ((lane & 16) ? 16 : 0);
  v16bf b;
#pragma unroll
  for (int e = 0; e < 16; ++e) b[e] = p[e];
  return b;
}

// ---------------------------------------------------------------------------
// Pre-pass: fp32 -> bf16 convert, and convert+transpose for weights.
// ---------------------------------------------------------------------------
__global__ void cvt_bf16_kernel(const float* __restrict__ in,
                                __bf16* __restrict__ out, int n) {
  int i = blockIdx.x * blockDim.x + threadIdx.x;
  if (i < n) out[i] = (__bf16)in[i];
}

// WT[n][k] = (bf16) W[k][n]; K, N multiples of 32. Block 32x8, tile 32x32.
__global__ __launch_bounds__(256) void transpose_bf16_kernel(
    const float* __restrict__ W, __bf16* __restrict__ WT, int K, int N) {
  __shared__ float tile[32][33];
  int k0 = blockIdx.x * 32, n0 = blockIdx.y * 32;
  int tx = threadIdx.x, ty = threadIdx.y;
#pragma unroll
  for (int i = 0; i < 32; i += 8)
    tile[ty + i][tx] = W[(size_t)(k0 + ty + i) * N + n0 + tx];
  __syncthreads();
#pragma unroll
  for (int i = 0; i < 32; i += 8)
    WT[(size_t)(n0 + ty + i) * K + k0 + tx] = (__bf16)tile[tx][ty + i];
}

// ---------------------------------------------------------------------------
// Double-buffered LDS-tiled WMMA GEMM: Out[M,N] = A[M,K] @ W[K,N], A bf16
// row-major, W given as WT[n][k] bf16. Block 256 threads = 8 waves (4M x 2N),
// block tile 128x128, K-step 64, wave tile 32x64 (2x4 WMMA per k-step).
// LDS rows padded to 72 elems (144 B): (36*row mod 64) spreads banks and
// chunk addresses stay 16B-aligned for the async b128 copies.
// ---------------------------------------------------------------------------
#define BM    128
#define BN    128
#define BK    64
#define LDT_  72   // padded LDS row stride in bf16 elements (144 B)

template <typename OutT, bool TRANS_OUT>
__global__ __launch_bounds__(256) void gemm_kernel(
    const __bf16* __restrict__ A, const __bf16* __restrict__ WT,
    OutT* __restrict__ Out, int M, int N, int K) {
  __shared__ __align__(16) __bf16 Abuf[2][BM * LDT_];  // 2 x 18,432 B
  __shared__ __align__(16) __bf16 Bbuf[2][BN * LDT_];  // 2 x 18,432 B

  int tid  = threadIdx.x;
  int lane = tid & 31;
  int wid  = tid >> 5;
  int nblk = N / BN;
  int bm0  = (blockIdx.x / nblk) * BM;
  int bn0  = (blockIdx.x % nblk) * BN;
  int wm   = (wid >> 1) * 32;  // 0,32,64,96
  int wn   = (wid & 1) * 64;   // 0,64

  v8f c[2][4];
#pragma unroll
  for (int i = 0; i < 2; ++i)
#pragma unroll
    for (int j = 0; j < 4; ++j)
#pragma unroll
      for (int r = 0; r < 8; ++r) c[i][j][r] = 0.0f;

  // Per-tile staging: A 128x64 bf16 = 1024 16B-chunks (4/thread), B likewise.
  auto stage = [&](int buf, int k0) {
#pragma unroll
    for (int i = 0; i < 4; ++i) {
      int cidx = i * 256 + tid;
      int row  = cidx >> 3;            // 8 chunks (of 8 elems) per row
      int c8   = (cidx & 7) * 8;
      cp_async16(&Abuf[buf][row * LDT_ + c8],
                 A + (size_t)(bm0 + row) * K + k0 + c8);
    }
#pragma unroll
    for (int i = 0; i < 4; ++i) {
      int cidx = i * 256 + tid;
      int row  = cidx >> 3;
      int c8   = (cidx & 7) * 8;
      cp_async16(&Bbuf[buf][row * LDT_ + c8],
                 WT + (size_t)(bn0 + row) * K + k0 + c8);
    }
  };

  int nk = K / BK;
  stage(0, 0);
  for (int t = 0; t < nk; ++t) {
    int cur = t & 1;
    if (t + 1 < nk) {
      stage(cur ^ 1, (t + 1) * BK);  // overlap next tile copies with compute
      cp_async_wait_keep8();         // current tile's copies have landed
    } else {
      cp_async_wait_all();
    }
    __syncthreads();

    const __bf16* Ab = Abuf[cur];
    const __bf16* Bb = Bbuf[cur];
#pragma unroll
    for (int ks = 0; ks < BK; ks += 32) {
      v16bf a0 = load_a_frag(Ab, LDT_, wm, ks, lane);
      v16bf a1 = load_a_frag(Ab, LDT_, wm + 16, ks, lane);
      v16bf bb[4];
#pragma unroll
      for (int j = 0; j < 4; ++j)
        bb[j] = load_b_frag_kfast(Bb, LDT_, wn + j * 16, ks, lane);
#pragma unroll
      for (int j = 0; j < 4; ++j) {
        c[0][j] = wmma_bf16(a0, bb[j], c[0][j]);
        c[1][j] = wmma_bf16(a1, bb[j], c[1][j]);
      }
    }
    __syncthreads();  // all waves done with `cur` before it is restaged
  }

  // D layout: lane L holds N = L&15; VGPR r holds M = r + 8*(L>=16)
  int lanehi8 = (lane & 16) ? 8 : 0;
  int n = lane & 15;
#pragma unroll
  for (int i = 0; i < 2; ++i)
#pragma unroll
    for (int j = 0; j < 4; ++j)
#pragma unroll
      for (int r = 0; r < 8; ++r) {
        int m   = bm0 + wm + i * 16 + r + lanehi8;
        int col = bn0 + wn + j * 16 + n;
        if (TRANS_OUT)
          Out[(size_t)col * M + m] = (OutT)c[i][j][r];  // V^T: 16B/lane runs
        else
          Out[(size_t)m * N + col] = (OutT)c[i][j][r];
      }
}

// ---------------------------------------------------------------------------
// RoPE (in place on bf16 qbuf, kbuf), fp32 math.
// ---------------------------------------------------------------------------
__global__ void rope_kernel(__bf16* __restrict__ qbuf, __bf16* __restrict__ kbuf) {
  const int QP = MTOT * (C_ / 2);
  const int KP = MTOT * (KVD_ / 2);
  int i = blockIdx.x * blockDim.x + threadIdx.x;
  __bf16* buf;
  int row, colp, ld;
  if (i < QP) {
    buf = qbuf; ld = C_;
    row = i / (C_ / 2); colp = i % (C_ / 2);
  } else {
    i -= QP;
    if (i >= KP) return;
    buf = kbuf; ld = KVD_;
    row = i / (KVD_ / 2); colp = i % (KVD_ / 2);
  }
  int col = colp * 2;
  int t = row & (T_ - 1);
  int d = col & (D_ - 1);  // even; inv_freq exponent = d/D
  float inv = __expf(-(float)d * (9.2103403719761836f / (float)D_));  // ln(1e4)
  float ang = (float)t * inv;
  float s, c;
  __sincosf(ang, &s, &c);
  size_t idx = (size_t)row * ld + col;
  float x1 = (float)buf[idx];
  float x2 = (float)buf[idx + 1];
  buf[idx]     = (__bf16)(x1 * c - x2 * s);
  buf[idx + 1] = (__bf16)(x2 * c + x1 * s);
}

// ---------------------------------------------------------------------------
// Flash attention: one wave per (b, h, 16-row query tile). q/k/v are bf16;
// k rows and vT rows are K-fast so every operand is a contiguous per-lane run.
// ---------------------------------------------------------------------------
__global__ __launch_bounds__(32) void attn_kernel(
    const __bf16* __restrict__ qbuf, const __bf16* __restrict__ kbuf,
    const __bf16* __restrict__ vT, __bf16* __restrict__ obuf) {
  __shared__ __bf16 pTile[16 * 32];  // P staged 16(q) x 32(keys)

  const int QT = T_ / 16;
  int lane = threadIdx.x;
  int bid = blockIdx.x;
  int qt = bid % QT;
  int h  = (bid / QT) % H_;
  int b  = bid / (QT * H_);
  int kvh = h / (H_ / KV_);
  int q0 = qt * 16;

  const __bf16* qbase = qbuf + (size_t)b * T_ * C_ + h * D_;     // ld = C_
  const __bf16* kbase = kbuf + (size_t)b * T_ * KVD_ + kvh * D_; // ld = KVD_
  // vT[dcol][m]: dcol in [0,512), m = b*T + t
  const __bf16* vbase = vT + (size_t)(kvh * D_) * MTOT + (size_t)b * T_;

  v16bf aq0 = load_a_frag(qbase, C_, q0, 0, lane);
  v16bf aq1 = load_a_frag(qbase, C_, q0, 32, lane);

  v8f o[4];
#pragma unroll
  for (int j = 0; j < 4; ++j)
#pragma unroll
    for (int r = 0; r < 8; ++r) o[j][r] = 0.0f;

  float mst[8], lst[8];
#pragma unroll
  for (int r = 0; r < 8; ++r) { mst[r] = -1e30f; lst[r] = 0.0f; }

  int lanehi8 = (lane & 16) ? 8 : 0;
  int ncol = lane & 15;
  const float scale = 0.125f;  // 1/sqrt(64)

  for (int kt = 0; kt < q0 + 16; kt += 32) {
    // ---- scores S = q @ k^T for 32 keys ----
    v8f s0, s1;
#pragma unroll
    for (int r = 0; r < 8; ++r) { s0[r] = 0.0f; s1[r] = 0.0f; }
    {
      v16bf bk;
      bk = load_b_frag_kfast(kbase, KVD_, kt, 0, lane);       s0 = wmma_bf16(aq0, bk, s0);
      bk = load_b_frag_kfast(kbase, KVD_, kt, 32, lane);      s0 = wmma_bf16(aq1, bk, s0);
      bk = load_b_frag_kfast(kbase, KVD_, kt + 16, 0, lane);  s1 = wmma_bf16(aq0, bk, s1);
      bk = load_b_frag_kfast(kbase, KVD_, kt + 16, 32, lane); s1 = wmma_bf16(aq1, bk, s1);
    }

    // ---- scale, causal mask, online softmax update ----
#pragma unroll
    for (int r = 0; r < 8; ++r) {
      int qrow = q0 + r + lanehi8;
      float v0 = s0[r] * scale;
      float v1 = s1[r] * scale;
      if (kt + ncol > qrow)      v0 = -1e30f;
      if (kt + 16 + ncol > qrow) v1 = -1e30f;

      float mx = fmaxf(v0, v1);
#pragma unroll
      for (int dm = 8; dm >= 1; dm >>= 1)
        mx = fmaxf(mx, __shfl_xor(mx, dm, 32));  // masks <16: halves independent
      float mnew  = fmaxf(mst[r], mx);
      float alpha = __expf(mst[r] - mnew);
      mst[r] = mnew;

      float p0 = __expf(v0 - mnew);
      float p1 = __expf(v1 - mnew);
      float sum = p0 + p1;
#pragma unroll
      for (int dm = 8; dm >= 1; dm >>= 1)
        sum += __shfl_xor(sum, dm, 32);
      lst[r] = lst[r] * alpha + sum;
#pragma unroll
      for (int j = 0; j < 4; ++j) o[j][r] *= alpha;

      pTile[(r + lanehi8) * 32 + ncol]      = (__bf16)p0;
      pTile[(r + lanehi8) * 32 + 16 + ncol] = (__bf16)p1;
    }
    __syncthreads();  // order pTile writes vs cross-lane reads (1 wave: cheap)

    // ---- P relayout (D layout -> A fragment) via LDS ----
    v16bf ap;
    {
      int m = lane & 15;
      int koff = (lane & 16) ? 16 : 0;
      const __bf16* pp = &pTile[m * 32 + koff];
#pragma unroll
      for (int e = 0; e < 8; ++e) { ap[e] = pp[e]; ap[e + 8] = pp[e + 8]; }
    }

    // ---- O += P @ V: B-frag from vT rows (K = keys is the fast axis) ----
#pragma unroll
    for (int j = 0; j < 4; ++j) {
      v16bf bv;
      {
        int n = j * 16 + ncol;
        const __bf16* p = vbase + (size_t)n * MTOT + kt + ((lane & 16) ? 16 : 0);
#pragma unroll
        for (int e = 0; e < 16; ++e) bv[e] = p[e];
      }
      o[j] = wmma_bf16(ap, bv, o[j]);
    }
    __syncthreads();
  }

  // ---- normalize and store (bf16) ----
  __bf16* ob = obuf + (size_t)b * T_ * C_ + h * D_;
#pragma unroll
  for (int r = 0; r < 8; ++r) {
    float inv_l = 1.0f / lst[r];
#pragma unroll
    for (int j = 0; j < 4; ++j)
      ob[(size_t)(q0 + r + lanehi8) * C_ + j * 16 + ncol] =
          (__bf16)(o[j][r] * inv_l);
  }
}

// ---------------------------------------------------------------------------
extern "C" void kernel_launch(void* const* d_in, const int* in_sizes, int n_in,
                              void* d_out, int out_size, void* d_ws,
                              size_t ws_size, hipStream_t stream) {
  const float* x  = (const float*)d_in[0];
  const float* Wq = (const float*)d_in[1];
  const float* Wk = (const float*)d_in[2];
  const float* Wv = (const float*)d_in[3];
  const float* Wp = (const float*)d_in[4];
  float* out = (float*)d_out;

  // Workspace layout in bf16 elements (total 38M bf16 = 76 MB).
  __bf16* wsb  = (__bf16*)d_ws;
  const size_t Mi = 1024 * 1024;
  __bf16* xbf  = wsb;             // 8M  : x          [4096][2048]
  __bf16* wqT  = xbf  + 8 * Mi;   // 4M  : Wq^T       [2048][2048]
  __bf16* wkT  = wqT  + 4 * Mi;   // 1M  : Wk^T       [512][2048]
  __bf16* wvT  = wkT  + 1 * Mi;   // 1M  : Wv^T       [512][2048]
  __bf16* wpT  = wvT  + 1 * Mi;   // 4M  : Wp^T       [2048][2048]
  __bf16* qbuf = wpT  + 4 * Mi;   // 8M  : q          [4096][2048]
  __bf16* kbuf = qbuf + 8 * Mi;   // 2M  : k          [4096][512]
  __bf16* vTb  = kbuf + 2 * Mi;   // 2M  : v^T        [512][4096]
  __bf16* obuf = vTb  + 2 * Mi;   // 8M  : attn out   [4096][2048]

  // ---- pre-pass: conversions (hoists all cvt out of GEMM hot loops) ----
  cvt_bf16_kernel<<<(MTOT * C_ + 255) / 256, 256, 0, stream>>>(x, xbf, MTOT * C_);
  {
    dim3 blk(32, 8);
    transpose_bf16_kernel<<<dim3(C_ / 32, C_ / 32), blk, 0, stream>>>(Wq, wqT, C_, C_);
    transpose_bf16_kernel<<<dim3(C_ / 32, KVD_ / 32), blk, 0, stream>>>(Wk, wkT, C_, KVD_);
    transpose_bf16_kernel<<<dim3(C_ / 32, KVD_ / 32), blk, 0, stream>>>(Wv, wvT, C_, KVD_);
    transpose_bf16_kernel<<<dim3(C_ / 32, C_ / 32), blk, 0, stream>>>(Wp, wpT, C_, C_);
  }

  // ---- QKV projections (V stored transposed) ----
  gemm_kernel<__bf16, false><<<(MTOT / BM) * (C_ / BN), 256, 0, stream>>>(
      xbf, wqT, qbuf, MTOT, C_, C_);
  gemm_kernel<__bf16, false><<<(MTOT / BM) * (KVD_ / BN), 256, 0, stream>>>(
      xbf, wkT, kbuf, MTOT, KVD_, C_);
  gemm_kernel<__bf16, true><<<(MTOT / BM) * (KVD_ / BN), 256, 0, stream>>>(
      xbf, wvT, vTb, MTOT, KVD_, C_);

  // ---- RoPE ----
  {
    int total = MTOT * (C_ / 2) + MTOT * (KVD_ / 2);
    rope_kernel<<<(total + 255) / 256, 256, 0, stream>>>(qbuf, kbuf);
  }

  // ---- causal GQA attention ----
  attn_kernel<<<B_ * H_ * (T_ / 16), 32, 0, stream>>>(qbuf, kbuf, vTb, obuf);

  // ---- output projection (fp32 out) ----
  gemm_kernel<float, false><<<(MTOT / BM) * (C_ / BN), 256, 0, stream>>>(
      obuf, wpT, out, MTOT, C_, C_);
}
